// NonLocal2d_39496519254186
// MI455X (gfx1250) — matrix operand order repaired and checked by reading, not compiled, not run on hardware
//
#include <hip/hip_runtime.h>
#include <hip/hip_bf16.h>

// Problem constants (from reference setup_inputs)
#define HW_  4096   // H*W = 64*64
#define C_   512
#define CI_  256
#define B_   4
#define EPS_ 1e-5f

typedef __attribute__((ext_vector_type(16))) __bf16 v16bf;
typedef __attribute__((ext_vector_type(8)))  float  v8f;

__device__ __forceinline__ __bf16 f2bf(float f) {
  union { float f; unsigned u; } v; v.f = f;
  unsigned r = v.u + 0x7FFFu + ((v.u >> 16) & 1u);   // round-to-nearest-even
  unsigned short h = (unsigned short)(r >> 16);
  return __builtin_bit_cast(__bf16, h);
}

union FragU { v16bf v; uint4 q[2]; };

// A-fragment (16x32 bf16), ISA layout: lane holds row m=lane&15;
// elems 0..7 -> K = (lane>=16?8:0)+j ; elems 8..15 -> K = 16+(lane>=16?8:0)+j
// A stored row-major [m][k] with leading dim lda. Two contiguous 16B loads.
__device__ __forceinline__ v16bf load_a_frag(const __bf16* A, int lda, int m0, int k0) {
  int lane = threadIdx.x & 31;
  int m  = m0 + (lane & 15);
  int kh = (lane >> 4) * 8;
  const __bf16* p = A + (size_t)m * lda + k0 + kh;
  FragU u;
  u.q[0] = *(const uint4*)(p);        // K = kh .. kh+7
  u.q[1] = *(const uint4*)(p + 16);   // K = 16+kh .. 16+kh+7
  return u.v;
}

// B-fragment (32x16 bf16): lane holds column n=lane&15; elems 0..7 -> K = (lane>=16?8:0)+j,
// elems 8..15 -> K = 16+(lane>=16?8:0)+j. B stored as [n][k] (column-contiguous-in-K),
// leading dim ldb -> two 16B loads per lane (K offsets 0 and 16).
__device__ __forceinline__ v16bf load_b_frag(const __bf16* Bnk, int ldb, int n0, int k0) {
  int lane = threadIdx.x & 31;
  int n = n0 + (lane & 15);
  int k = k0 + (lane >> 4) * 8;
  const uint4* p = (const uint4*)(Bnk + (size_t)n * ldb + k);
  FragU u;
  u.q[0] = p[0];
  u.q[1] = p[2];
  return u.v;
}

__device__ __forceinline__ v8f wmma_bf16(v16bf a, v16bf b, v8f c) {
  return __builtin_amdgcn_wmma_f32_16x16x32_bf16(false, a, false, b, (short)0, c,
                                                 false, false);
}

// ---------------------------------------------------------------- converts
__global__ void w2bf_kernel(const float* __restrict__ src, __bf16* __restrict__ dst, int n) {
  int i = blockIdx.x * blockDim.x + threadIdx.x;
  if (i < n) dst[i] = f2bf(src[i]);
}

// x (B,C,HW) f32  ->  XT (B,HW,C) bf16   (token-major for B-fragment loads)
__global__ void xt_kernel(const float* __restrict__ x, __bf16* __restrict__ XT) {
  size_t i = (size_t)blockIdx.x * blockDim.x + threadIdx.x;  // (b,n,c), c fastest
  int c = (int)(i % C_);
  size_t t = i / C_;
  int n = (int)(t % HW_);
  int b = (int)(t / HW_);
  XT[i] = f2bf(x[((size_t)b * C_ + c) * HW_ + n]);
}

// --------------------------------------------------- fused 3-way projection
// out[ci][n] = sum_c W[ci][c] * x[c][n] + bias[ci]
// grid: (HW/64, CI/64, B*3), block 128 (4 waves; wave w owns 16 rows of m)
__global__ void proj_kernel(const __bf16* __restrict__ XT,
                            const __bf16* __restrict__ Wg, const __bf16* __restrict__ Wt,
                            const __bf16* __restrict__ Wp,
                            const float* __restrict__ gb, const float* __restrict__ tb,
                            const float* __restrict__ pb,
                            __bf16* __restrict__ G,   // (B, CI, HW) ci-major
                            __bf16* __restrict__ TH,  // (B, HW, CI) token-major
                            __bf16* __restrict__ PH)  // (B, HW, CI) token-major
{
  int nb = blockIdx.x, mb = blockIdx.y, z = blockIdx.z;
  int b = z / 3, proj = z % 3;
  int wave = threadIdx.x >> 5, lane = threadIdx.x & 31;
  const __bf16* W    = (proj == 0) ? Wg : (proj == 1) ? Wt : Wp;
  const float*  bias = (proj == 0) ? gb : (proj == 1) ? tb : pb;
  const __bf16* Bmat = XT + (size_t)b * HW_ * C_;    // [token][c]
  int m0 = mb * 64 + wave * 16;
  int n0 = nb * 64;

  v8f acc[4] = {v8f{}, v8f{}, v8f{}, v8f{}};
  for (int k = 0; k < C_; k += 32) {
    v16bf a = load_a_frag(W, C_, m0, k);
#pragma unroll
    for (int j = 0; j < 4; ++j) {
      v16bf bf = load_b_frag(Bmat, C_, n0 + j * 16, k);
      acc[j] = wmma_bf16(a, bf, acc[j]);
    }
  }
  int half = lane >> 4, nl = lane & 15;
#pragma unroll
  for (int j = 0; j < 4; ++j) {
#pragma unroll
    for (int v = 0; v < 8; ++v) {
      int m = m0 + v + 8 * half;
      int n = n0 + j * 16 + nl;
      float val = acc[j][v] + bias[m];
      if (proj == 0)      G [((size_t)b * CI_ + m) * HW_ + n] = f2bf(val);
      else if (proj == 1) TH[((size_t)b * HW_ + n) * CI_ + m] = f2bf(val);
      else                PH[((size_t)b * HW_ + n) * CI_ + m] = f2bf(val);
    }
  }
}

// ------------------------------------------------------- flash attention
// y[q][ci] = softmax_k( theta[q]·phi[k] / 16 ) · g[k][ci]
// grid: (HW/64, B), block 128 (4 waves; wave owns 16 query rows)
__global__ void flash_kernel(const __bf16* __restrict__ TH,  // (B,HW,CI)
                             const __bf16* __restrict__ PH,  // (B,HW,CI)
                             const __bf16* __restrict__ G,   // (B,CI,HW)
                             __bf16* __restrict__ Y)         // (B,HW,CI)
{
  __shared__ __bf16 plds[4][16][32];   // per-wave P-tile bounce (C-layout -> A-layout)
  int b = blockIdx.y;
  int wave = threadIdx.x >> 5, lane = threadIdx.x & 31;
  int q0 = blockIdx.x * 64 + wave * 16;
  const __bf16* thb = TH + (size_t)b * HW_ * CI_;
  const __bf16* phb = PH + (size_t)b * HW_ * CI_;
  const __bf16* gb  = G  + (size_t)b * CI_ * HW_;

  // Pre-load Q fragments (16 rows x full CI) — reused for all 128 key blocks
  v16bf qa[8];
#pragma unroll
  for (int ks = 0; ks < 8; ++ks) qa[ks] = load_a_frag(thb, CI_, q0, ks * 32);

  v8f acc[16];
#pragma unroll
  for (int j = 0; j < 16; ++j) acc[j] = v8f{};
  float mrow[8], lrow[8];
#pragma unroll
  for (int v = 0; v < 8; ++v) { mrow[v] = -1e30f; lrow[v] = 0.0f; }

  const float rs = 0.0625f;   // 1/sqrt(CI) = 1/16
  int half = lane >> 4, nl = lane & 15;

  for (int kb = 0; kb < HW_; kb += 32) {
    if (kb + 32 < HW_) {
      __builtin_prefetch(phb + (size_t)(kb + 32) * CI_, 0, 1);
      __builtin_prefetch(gb + (size_t)(kb + 32), 0, 1);
    }
    // S = Q · K^T  (two 16x16 tiles covering 32 keys)
    v8f s0 = v8f{}, s1 = v8f{};
#pragma unroll
    for (int ks = 0; ks < 8; ++ks) {
      v16bf b0 = load_b_frag(phb, CI_, kb,      ks * 32);
      v16bf b1 = load_b_frag(phb, CI_, kb + 16, ks * 32);
      s0 = wmma_bf16(qa[ks], b0, s0);
      s1 = wmma_bf16(qa[ks], b1, s1);
    }
    // online softmax (row = v + 8*half, replicated over 16 lanes of the half)
    float p0[8], p1[8], corr[8];
#pragma unroll
    for (int v = 0; v < 8; ++v) {
      float a0 = s0[v] * rs, a1 = s1[v] * rs;
      float cur = fmaxf(a0, a1);
#pragma unroll
      for (int off = 1; off < 16; off <<= 1)
        cur = fmaxf(cur, __shfl_xor(cur, off, 32));
      float mnew = fmaxf(mrow[v], cur);
      corr[v] = __expf(mrow[v] - mnew);
      mrow[v] = mnew;
      p0[v] = __expf(a0 - mnew);
      p1[v] = __expf(a1 - mnew);
      float rsum = p0[v] + p1[v];
#pragma unroll
      for (int off = 1; off < 16; off <<= 1)
        rsum += __shfl_xor(rsum, off, 32);
      lrow[v] = lrow[v] * corr[v] + rsum;
    }
    // rescale running output
#pragma unroll
    for (int j = 0; j < 16; ++j)
#pragma unroll
      for (int v = 0; v < 8; ++v) acc[j][v] *= corr[v];
    // bounce P (C-layout f32) -> LDS -> A-fragment (bf16)
#pragma unroll
    for (int v = 0; v < 8; ++v) {
      int row = v + 8 * half;
      plds[wave][row][nl]      = f2bf(p0[v]);
      plds[wave][row][16 + nl] = f2bf(p1[v]);
    }
    v16bf pa;
    {
      int m = lane & 15, kh = (lane >> 4) * 8;
      const __bf16* base = &plds[wave][0][0];
      FragU u;
      u.q[0] = *(const uint4*)(base + m * 32 + kh);
      u.q[1] = *(const uint4*)(base + m * 32 + 16 + kh);
      pa = u.v;
    }
    // O += P · G  (16 n-tiles over CI=256)
#pragma unroll
    for (int j = 0; j < 16; ++j) {
      v16bf bg = load_b_frag(gb, HW_, j * 16, kb);
      acc[j] = wmma_bf16(pa, bg, acc[j]);
    }
  }
  // finalize: divide by row sums, store token-major bf16
  float inv[8];
#pragma unroll
  for (int v = 0; v < 8; ++v) inv[v] = 1.0f / lrow[v];
#pragma unroll
  for (int j = 0; j < 16; ++j) {
#pragma unroll
    for (int v = 0; v < 8; ++v) {
      int q  = q0 + v + 8 * half;
      int ci = j * 16 + nl;
      Y[((size_t)b * HW_ + q) * CI_ + ci] = f2bf(acc[j][v] * inv[v]);
    }
  }
}

// ------------------------------------------------------- output 1x1 conv
// z[co][n] = sum_ci WW[co][ci] * y[n][ci] + Wb[co]   (f32 out for BN stats)
// grid: (HW/64, C/64, B), block 128
__global__ void convout_kernel(const __bf16* __restrict__ WW,  // (C, CI)
                               const float* __restrict__ Wb,
                               const __bf16* __restrict__ Y,   // (B,HW,CI)
                               float* __restrict__ Z)          // (B,C,HW)
{
  int nb = blockIdx.x, mb = blockIdx.y, b = blockIdx.z;
  int wave = threadIdx.x >> 5, lane = threadIdx.x & 31;
  const __bf16* Bmat = Y + (size_t)b * HW_ * CI_;
  int m0 = mb * 64 + wave * 16;
  int n0 = nb * 64;
  v8f acc[4] = {v8f{}, v8f{}, v8f{}, v8f{}};
  for (int k = 0; k < CI_; k += 32) {
    v16bf a = load_a_frag(WW, CI_, m0, k);
#pragma unroll
    for (int j = 0; j < 4; ++j) {
      v16bf bf = load_b_frag(Bmat, CI_, n0 + j * 16, k);
      acc[j] = wmma_bf16(a, bf, acc[j]);
    }
  }
  int half = lane >> 4, nl = lane & 15;
#pragma unroll
  for (int j = 0; j < 4; ++j)
#pragma unroll
    for (int v = 0; v < 8; ++v) {
      int m = m0 + v + 8 * half;
      int n = n0 + j * 16 + nl;
      Z[((size_t)b * C_ + m) * HW_ + n] = acc[j][v] + Wb[m];
    }
}

// ------------------------------------------------------- BN statistics
__global__ void bnstat_kernel(const float* __restrict__ Z,
                              float* __restrict__ mean, float* __restrict__ var) {
  __shared__ float sh0[256], sh1[256];
  int c = blockIdx.x, tid = threadIdx.x;
  float s = 0.0f, ss = 0.0f;
  for (int b = 0; b < B_; ++b) {
    const float* p = Z + ((size_t)b * C_ + c) * HW_;
    for (int n = tid; n < HW_; n += 256) { float v = p[n]; s += v; ss += v * v; }
  }
  sh0[tid] = s; sh1[tid] = ss;
  __syncthreads();
  for (int o = 128; o > 0; o >>= 1) {
    if (tid < o) { sh0[tid] += sh0[tid + o]; sh1[tid] += sh1[tid + o]; }
    __syncthreads();
  }
  if (tid == 0) {
    float m = sh0[0] * (1.0f / (B_ * HW_));
    mean[c] = m;
    var[c]  = sh1[0] * (1.0f / (B_ * HW_)) - m * m;
  }
}

// -------------------------------------------- BN apply + residual (float4)
__global__ void bnapply_kernel(const float* __restrict__ Z, const float* __restrict__ x,
                               const float* __restrict__ gamma, const float* __restrict__ beta,
                               const float* __restrict__ mean, const float* __restrict__ var,
                               float* __restrict__ out) {
  size_t i4 = (size_t)blockIdx.x * blockDim.x + threadIdx.x;
  size_t i = i4 * 4;
  int c = (int)((i / HW_) % C_);   // HW_ % 4 == 0, so c constant within vec4
  float g = gamma[c], bt = beta[c];
  float m = mean[c];
  float rinv = rsqrtf(var[c] + EPS_);
  float4 z = *(const float4*)(Z + i);
  float4 xr = *(const float4*)(x + i);
  float4 o;
  o.x = g * (z.x - m) * rinv + bt + xr.x;
  o.y = g * (z.y - m) * rinv + bt + xr.y;
  o.z = g * (z.z - m) * rinv + bt + xr.z;
  o.w = g * (z.w - m) * rinv + bt + xr.w;
  *(float4*)(out + i) = o;
}

// ---------------------------------------------------------------- launch
extern "C" void kernel_launch(void* const* d_in, const int* in_sizes, int n_in,
                              void* d_out, int out_size, void* d_ws, size_t ws_size,
                              hipStream_t stream) {
  const float* x       = (const float*)d_in[0];
  const float* g_w     = (const float*)d_in[1];
  const float* g_b     = (const float*)d_in[2];
  const float* theta_w = (const float*)d_in[3];
  const float* theta_b = (const float*)d_in[4];
  const float* phi_w   = (const float*)d_in[5];
  const float* phi_b   = (const float*)d_in[6];
  const float* W_w     = (const float*)d_in[7];
  const float* W_b     = (const float*)d_in[8];
  const float* bn_g    = (const float*)d_in[9];
  const float* bn_b    = (const float*)d_in[10];
  float* out = (float*)d_out;

  // workspace carve-up (256B aligned)
  char* ws = (char*)d_ws;
  size_t off = 0;
  auto alloc = [&](size_t bytes) { char* p = ws + off; off = (off + bytes + 255) & ~(size_t)255; return p; };
  __bf16* XT  = (__bf16*)alloc((size_t)B_ * HW_ * C_ * 2);   // (B,HW,C)
  __bf16* WGb = (__bf16*)alloc((size_t)CI_ * C_ * 2);
  __bf16* WTb = (__bf16*)alloc((size_t)CI_ * C_ * 2);
  __bf16* WPb = (__bf16*)alloc((size_t)CI_ * C_ * 2);
  __bf16* WWb = (__bf16*)alloc((size_t)C_ * CI_ * 2);
  __bf16* G   = (__bf16*)alloc((size_t)B_ * CI_ * HW_ * 2);  // (B,CI,HW)
  __bf16* TH  = (__bf16*)alloc((size_t)B_ * HW_ * CI_ * 2);  // (B,HW,CI)
  __bf16* PH  = (__bf16*)alloc((size_t)B_ * HW_ * CI_ * 2);  // (B,HW,CI)
  __bf16* Y   = (__bf16*)alloc((size_t)B_ * HW_ * CI_ * 2);  // (B,HW,CI)
  float*  Z   = (float*)alloc((size_t)B_ * C_ * HW_ * 4);    // (B,C,HW)
  float*  MEAN = (float*)alloc((size_t)C_ * 4);
  float*  VAR  = (float*)alloc((size_t)C_ * 4);
  (void)ws_size; (void)in_sizes; (void)n_in; (void)out_size;

  // 1. weight + activation conversion to bf16
  int wn = CI_ * C_;
  w2bf_kernel<<<(wn + 255) / 256, 256, 0, stream>>>(g_w, WGb, wn);
  w2bf_kernel<<<(wn + 255) / 256, 256, 0, stream>>>(theta_w, WTb, wn);
  w2bf_kernel<<<(wn + 255) / 256, 256, 0, stream>>>(phi_w, WPb, wn);
  w2bf_kernel<<<(wn + 255) / 256, 256, 0, stream>>>(W_w, WWb, wn);
  xt_kernel<<<(B_ * HW_ * C_) / 256, 256, 0, stream>>>(x, XT);

  // 2. fused g/theta/phi projections (WMMA GEMM)
  proj_kernel<<<dim3(HW_ / 64, CI_ / 64, B_ * 3), 128, 0, stream>>>(
      XT, WGb, WTb, WPb, g_b, theta_b, phi_b, G, TH, PH);

  // 3. flash attention (WMMA, online softmax, never materializes 4096x4096)
  flash_kernel<<<dim3(HW_ / 64, B_), 128, 0, stream>>>(TH, PH, G, Y);

  // 4. output 1x1 conv (WMMA GEMM, f32 result for BN)
  convout_kernel<<<dim3(HW_ / 64, C_ / 64, B_), 128, 0, stream>>>(WWb, W_b, Y, Z);

  // 5. BN batch statistics + 6. BN apply + residual
  bnstat_kernel<<<C_, 256, 0, stream>>>(Z, MEAN, VAR);
  bnapply_kernel<<<(B_ * C_ * HW_ / 4) / 256, 256, 0, stream>>>(
      Z, x, bn_g, bn_b, MEAN, VAR, out);
}
